// VSDF_60043642798803
// MI455X (gfx1250) — matrix operand order repaired and checked by reading, not compiled, not run on hardware
//
#include <hip/hip_runtime.h>
#include <hip/hip_bf16.h>
#include <math.h>

// ---------------------------------------------------------------------------
// MI455X / gfx1250 implementation. wave32; all GEMM-shaped math lowered to
// v_wmma_f32_16x16x32_bf16 (fp32 accumulate). Attention is flash-style.
// GEMM tiles staged via GLOBAL_LOAD_ASYNC_TO_LDS (ASYNCcnt) when available.
// ---------------------------------------------------------------------------

typedef __attribute__((ext_vector_type(16))) __bf16 v16bf;
typedef __attribute__((ext_vector_type(8)))  float  v8f;

#define DIM     512
#define NPATCH  1024
#define NHEADS  8
#define HD      64
#define TD      1536          // 3*DIM
#define MROWS   (8 * NPATCH)  // 8192

#if defined(__AMDGCN__) && __has_builtin(__builtin_amdgcn_global_load_async_to_lds_b128)
#define HAVE_ASYNC_LDS 1
#endif

static __device__ inline float gelu_f(float x) {
    return 0.5f * x * (1.0f + erff(x * 0.70710678118654752f));
}

static __device__ inline v8f wmma_bf16(v16bf a, v16bf b, v8f c) {
    // (neg_a, A, neg_b, B, c_mod, C, reuse_a, reuse_b)
    return __builtin_amdgcn_wmma_f32_16x16x32_bf16(false, a, false, b,
                                                   (short)0, c, false, false);
}

#ifdef HAVE_ASYNC_LDS
#define AS1 __attribute__((address_space(1)))
#define AS3 __attribute__((address_space(3)))
typedef int v4i_gcc __attribute__((vector_size(4 * sizeof(int))));
static __device__ inline void async_cp16(const float* g, float* l) {
    // 16-byte global -> LDS async copy, tracked by ASYNCcnt (no VGPR data path)
    __builtin_amdgcn_global_load_async_to_lds_b128(
        (AS1 v4i_gcc*)g, (AS3 v4i_gcc*)l, 0, 0);
}
#endif

// ---- fragment loaders (contiguous k = 16h + i order, wave32) ------------------------
// A (16x32 bf16), source row-major s[m*ld + k]
static __device__ inline v16bf lds_a_frag(const __bf16* s, int ld) {
    const int lane = threadIdx.x & 31;
    const int m  = lane & 15;
    const int kb = (lane & 16) ? 16 : 0;  // lanes 0-15: K 0-15; 16-31: K 16-31
    v16bf a;
#pragma unroll
    for (int i = 0; i < 16; ++i) {
        const int k = kb + i;
        a[i] = s[m * ld + k];
    }
    return a;
}
// B (32x16 bf16), source stored n-major: s[n*ld + k]
static __device__ inline v16bf lds_bn_frag(const __bf16* s, int ld) {
    const int lane = threadIdx.x & 31;
    const int n  = lane & 15;
    const int kb = (lane & 16) ? 16 : 0;  // lanes 0-15: K 0-15; 16-31: K 16-31
    v16bf b;
#pragma unroll
    for (int i = 0; i < 16; ++i) b[i] = s[n * ld + kb + i];
    return b;
}
// B (32x16 bf16), source stored k-major: s[k*ld + n]
static __device__ inline v16bf lds_bk_frag(const __bf16* s, int ld) {
    const int lane = threadIdx.x & 31;
    const int n  = lane & 15;
    const int kb = (lane & 16) ? 16 : 0;
    v16bf b;
#pragma unroll
    for (int i = 0; i < 16; ++i) b[i] = s[(kb + i) * ld + n];
    return b;
}
#ifdef HAVE_ASYNC_LDS
// f32-sourced variants (async-staged tiles): convert at gather time
static __device__ inline v16bf ldsf_a_frag(const float* s, int ld) {
    const int lane = threadIdx.x & 31;
    const int m  = lane & 15;
    const int kb = (lane & 16) ? 16 : 0;
    v16bf a;
#pragma unroll
    for (int i = 0; i < 16; ++i) {
        const int k = kb + i;
        a[i] = (__bf16)s[m * ld + k];
    }
    return a;
}
static __device__ inline v16bf ldsf_bk_frag(const float* s, int ld) {
    const int lane = threadIdx.x & 31;
    const int n  = lane & 15;
    const int kb = (lane & 16) ? 16 : 0;
    v16bf b;
#pragma unroll
    for (int i = 0; i < 16; ++i) b[i] = (__bf16)s[(kb + i) * ld + n];
    return b;
}
#endif

// ===========================================================================
// Generic WMMA GEMM: C[M,Nc] = act(A[M,K] @ W[K,Nc] + bias)
// block tile 128x128, 256 threads (8 waves x 16 rows), K-step 32.
// EPI: 0 = bias, 1 = bias + exact GELU
// ===========================================================================
#define ALD  40   // padded bf16 stride (fallback path)
#define ALDF 36   // padded f32 stride, A tile rows (async path, 16B-chunk aligned)
#define BLDF 132  // padded f32 stride, W tile rows (async path)

template<int EPI>
__global__ __launch_bounds__(256)
void k_gemm(const float* __restrict__ A, const float* __restrict__ W,
            const float* __restrict__ bias, float* __restrict__ C,
            int K, int Nc)
{
    const int n0   = blockIdx.x * 128;
    const int m0   = blockIdx.y * 128;
    const int tid  = threadIdx.x;
    const int wave = tid >> 5;
    const int lane = tid & 31;

    v8f acc[8];
#pragma unroll
    for (int c = 0; c < 8; ++c)
#pragma unroll
        for (int r = 0; r < 8; ++r) acc[c][r] = 0.0f;

#ifdef HAVE_ASYNC_LDS
    // ---- async staging path: raw f32 tiles via GLOBAL_LOAD_ASYNC_TO_LDS ----
    __shared__ float a_smf[128 * ALDF];   // [row][k]   (row-major, 16B row pad)
    __shared__ float b_smf[32 * BLDF];    // [k][n]     (k-major, 16B row pad)

    const int ar = tid >> 1;            // 0..127  A row
    const int ao = (tid & 1) * 16;      // 0/16    A col base (4 x b128)
    const int bk = tid >> 3;            // 0..31   W k row
    const int bo = (tid & 7) * 16;      // 0..112  W n base  (4 x b128)

    for (int k0 = 0; k0 < K; k0 += 32) {
        __syncthreads();   // previous tiles fully consumed
        {
            const float* ag = A + (size_t)(m0 + ar) * K + k0 + ao;
            float*       al = a_smf + ar * ALDF + ao;
            const float* wg = W + (size_t)(k0 + bk) * Nc + n0 + bo;
            float*       wl = b_smf + bk * BLDF + bo;
#pragma unroll
            for (int q = 0; q < 4; ++q) async_cp16(ag + q * 4, al + q * 4);
#pragma unroll
            for (int q = 0; q < 4; ++q) async_cp16(wg + q * 4, wl + q * 4);
        }
        if (k0 + 32 < K) {
            __builtin_prefetch(A + (size_t)(m0 + ar) * K + k0 + 32 + ao, 0, 1);
            __builtin_prefetch(W + (size_t)(k0 + 32 + bk) * Nc + n0 + bo, 0, 1);
        }
        asm volatile("s_wait_asynccnt 0x0" ::: "memory");
        __syncthreads();

        const v16bf af = ldsf_a_frag(a_smf + (wave * 16) * ALDF, ALDF);
#pragma unroll
        for (int c = 0; c < 8; ++c) {
            const v16bf bf_ = ldsf_bk_frag(b_smf + c * 16, BLDF);
            acc[c] = wmma_bf16(af, bf_, acc[c]);
        }
    }
#else
    // ---- fallback: VGPR staging with bf16 conversion before LDS ----
    __shared__ __bf16 a_sm[128 * ALD];   // [row][k]
    __shared__ __bf16 b_sm[128 * ALD];   // [n][k]

    const int ar = tid >> 1;            // 0..127  (A stage row)
    const int ac = (tid & 1) * 16;      // 0/16    (A stage col base)
    const int bk = tid >> 3;            // 0..31   (W stage k)
    const int bn = (tid & 7) * 16;      // 0..112  (W stage n base)

    for (int k0 = 0; k0 < K; k0 += 32) {
        __syncthreads();
        {
            const float4* ap = reinterpret_cast<const float4*>(
                A + (size_t)(m0 + ar) * K + k0 + ac);
#pragma unroll
            for (int q = 0; q < 4; ++q) {
                float4 f = ap[q];
                __bf16* d = a_sm + ar * ALD + ac + q * 4;
                d[0] = (__bf16)f.x; d[1] = (__bf16)f.y;
                d[2] = (__bf16)f.z; d[3] = (__bf16)f.w;
            }
        }
        {
            const float4* wp = reinterpret_cast<const float4*>(
                W + (size_t)(k0 + bk) * Nc + n0 + bn);
#pragma unroll
            for (int q = 0; q < 4; ++q) {
                float4 f = wp[q];
                b_sm[(bn + q * 4 + 0) * ALD + bk] = (__bf16)f.x;
                b_sm[(bn + q * 4 + 1) * ALD + bk] = (__bf16)f.y;
                b_sm[(bn + q * 4 + 2) * ALD + bk] = (__bf16)f.z;
                b_sm[(bn + q * 4 + 3) * ALD + bk] = (__bf16)f.w;
            }
        }
        if (k0 + 32 < K) {
            __builtin_prefetch(A + (size_t)(m0 + ar) * K + k0 + 32 + ac, 0, 1);
            __builtin_prefetch(W + (size_t)(k0 + 32 + bk) * Nc + n0 + bn, 0, 1);
        }
        __syncthreads();

        const v16bf af = lds_a_frag(a_sm + (wave * 16) * ALD, ALD);
#pragma unroll
        for (int c = 0; c < 8; ++c) {
            const v16bf bf_ = lds_bn_frag(b_sm + (c * 16) * ALD, ALD);
            acc[c] = wmma_bf16(af, bf_, acc[c]);
        }
    }
#endif

    // ---- epilogue ----
    const int n_l = lane & 15;
    const int mb  = (lane & 16) ? 8 : 0;
#pragma unroll
    for (int c = 0; c < 8; ++c) {
        const int col = n0 + c * 16 + n_l;
        const float bv = bias[col];
#pragma unroll
        for (int r = 0; r < 8; ++r) {
            float v = acc[c][r] + bv;
            if (EPI == 1) v = gelu_f(v);
            C[(size_t)(m0 + wave * 16 + mb + r) * Nc + col] = v;
        }
    }
}

// ===========================================================================
// Row LayerNorm over `cols` (in-place). One block per row.
// ===========================================================================
__global__ __launch_bounds__(256)
void k_ln_rows(float* __restrict__ Y, const float* __restrict__ g,
               const float* __restrict__ b, int cols)
{
    __shared__ float red[16];
    float* y = Y + (size_t)blockIdx.x * cols;
    const int tid = threadIdx.x, wave = tid >> 5, lane = tid & 31;

    float s = 0.f, s2 = 0.f;
    for (int c = tid; c < cols; c += 256) { float v = y[c]; s += v; s2 += v * v; }
#pragma unroll
    for (int off = 1; off < 32; off <<= 1) {
        s  += __shfl_xor(s,  off, 32);
        s2 += __shfl_xor(s2, off, 32);
    }
    if (lane == 0) { red[wave] = s; red[8 + wave] = s2; }
    __syncthreads();
    s = 0.f; s2 = 0.f;
#pragma unroll
    for (int w = 0; w < 8; ++w) { s += red[w]; s2 += red[8 + w]; }
    const float inv_n = 1.0f / (float)cols;
    const float mu  = s * inv_n;
    const float var = fmaxf(s2 * inv_n - mu * mu, 0.0f);
    const float rs  = rsqrtf(var + 1e-5f);
    for (int c = tid; c < cols; c += 256)
        y[c] = (y[c] - mu) * rs * g[c] + b[c];
}

// ===========================================================================
// Flash cosine attention. qkv: [B*N, 1536] (LayerNormed). out: [B*N, 512].
// One block = (b, h, 128 q rows); 8 waves x 16 q rows each.
// ===========================================================================
#define QLD 72
#define KLD 72
#define PLD 40

__global__ __launch_bounds__(256)
void k_attn(const float* __restrict__ qkv, const float* __restrict__ temperature,
            float* __restrict__ outp)
{
    __shared__ __bf16 q_sm[8 * 16 * QLD];
    __shared__ __bf16 k_sm[32 * KLD];
    __shared__ __bf16 v_sm[32 * KLD];
    __shared__ __bf16 p_sm[8 * 16 * PLD];

    const int qblk = blockIdx.x & 7;
    const int bh   = blockIdx.x >> 3;
    const int h    = bh & 7;
    const int b    = bh >> 3;
    const int q0   = qblk * 128;
    const int wave = threadIdx.x >> 5;
    const int lane = threadIdx.x & 31;
    const float temp = temperature[h];

    // ---- stage + L2-normalize Q (wave-private: 16 rows x 64) ----
    {
        const int row  = lane & 15;
        const int half = lane >> 4;
        const size_t tok = (size_t)(b * NPATCH + q0 + wave * 16 + row);
        const float4* s4 = reinterpret_cast<const float4*>(
            qkv + tok * TD + h * HD + half * 32);
        float t[32]; float ss = 0.f;
#pragma unroll
        for (int q = 0; q < 8; ++q) {
            float4 f = s4[q];
            t[4*q+0] = f.x; t[4*q+1] = f.y; t[4*q+2] = f.z; t[4*q+3] = f.w;
            ss += f.x*f.x + f.y*f.y + f.z*f.z + f.w*f.w;
        }
        ss += __shfl_xor(ss, 16, 32);
        const float inv = 1.0f / fmaxf(sqrtf(ss), 1e-12f);
        __bf16* qd = q_sm + (wave * 16 + row) * QLD + half * 32;
#pragma unroll
        for (int i = 0; i < 32; ++i) qd[i] = (__bf16)(t[i] * inv);
    }
    asm volatile("s_wait_dscnt 0x0" ::: "memory");
    const v16bf qa0 = lds_a_frag(q_sm + (wave * 16) * QLD, QLD);
    const v16bf qa1 = lds_a_frag(q_sm + (wave * 16) * QLD + 32, QLD);

    float m_i[8], l_i[8];
    v8f o[4];
#pragma unroll
    for (int r = 0; r < 8; ++r) { m_i[r] = -1e30f; l_i[r] = 0.f; }
#pragma unroll
    for (int c = 0; c < 4; ++c)
#pragma unroll
        for (int r = 0; r < 8; ++r) o[c][r] = 0.f;

    const int kr = threadIdx.x >> 3;        // 0..31 kv row
    const int ks = (threadIdx.x & 7) * 8;   // 8-col segment

    for (int j0 = 0; j0 < NPATCH; j0 += 32) {
        __syncthreads();
        // ---- stage K (L2-normalized) and V chunk (32 x 64) ----
        {
            const size_t tok = (size_t)(b * NPATCH + j0 + kr);
            const float4* k4 = reinterpret_cast<const float4*>(
                qkv + tok * TD + DIM + h * HD + ks);
            const float4* v4 = reinterpret_cast<const float4*>(
                qkv + tok * TD + 2 * DIM + h * HD + ks);
            float kv[8]; float ss = 0.f;
#pragma unroll
            for (int q = 0; q < 2; ++q) {
                float4 f = k4[q];
                kv[4*q+0] = f.x; kv[4*q+1] = f.y; kv[4*q+2] = f.z; kv[4*q+3] = f.w;
                ss += f.x*f.x + f.y*f.y + f.z*f.z + f.w*f.w;
            }
            ss += __shfl_xor(ss, 1, 8);
            ss += __shfl_xor(ss, 2, 8);
            ss += __shfl_xor(ss, 4, 8);
            const float inv = 1.0f / fmaxf(sqrtf(ss), 1e-12f);
#pragma unroll
            for (int i = 0; i < 8; ++i)
                k_sm[kr * KLD + ks + i] = (__bf16)(kv[i] * inv);
#pragma unroll
            for (int q = 0; q < 2; ++q) {
                float4 f = v4[q];
                v_sm[kr * KLD + ks + 4*q + 0] = (__bf16)f.x;
                v_sm[kr * KLD + ks + 4*q + 1] = (__bf16)f.y;
                v_sm[kr * KLD + ks + 4*q + 2] = (__bf16)f.z;
                v_sm[kr * KLD + ks + 4*q + 3] = (__bf16)f.w;
            }
        }
        __syncthreads();

        // ---- S(16x32) = Qhat @ Khat^T ----
        v8f s0, s1;
#pragma unroll
        for (int r = 0; r < 8; ++r) { s0[r] = 0.f; s1[r] = 0.f; }
        s0 = wmma_bf16(qa0, lds_bn_frag(k_sm, KLD), s0);
        s0 = wmma_bf16(qa1, lds_bn_frag(k_sm + 32, KLD), s0);
        s1 = wmma_bf16(qa0, lds_bn_frag(k_sm + 16 * KLD, KLD), s1);
        s1 = wmma_bf16(qa1, lds_bn_frag(k_sm + 16 * KLD + 32, KLD), s1);

        // ---- online softmax (per-row, cross-lane reductions width 16) ----
#pragma unroll
        for (int r = 0; r < 8; ++r) {
            float v0 = s0[r] * temp, v1 = s1[r] * temp;
            float mx = fmaxf(v0, v1);
#pragma unroll
            for (int off = 1; off < 16; off <<= 1)
                mx = fmaxf(mx, __shfl_xor(mx, off, 16));
            const float mnew = fmaxf(m_i[r], mx);
            const float corr = __expf(m_i[r] - mnew);
            const float p0 = __expf(v0 - mnew);
            const float p1 = __expf(v1 - mnew);
            float rs = p0 + p1;
#pragma unroll
            for (int off = 1; off < 16; off <<= 1)
                rs += __shfl_xor(rs, off, 16);
            l_i[r] = l_i[r] * corr + rs;
            m_i[r] = mnew;
            s0[r] = p0; s1[r] = p1;
            o[0][r] *= corr; o[1][r] *= corr; o[2][r] *= corr; o[3][r] *= corr;
        }

        // ---- C-layout -> A-layout via wave-private LDS, then O += P @ V ----
        {
            __bf16* pw = p_sm + (wave * 16) * PLD;
            const int n  = lane & 15;
            const int mb = (lane & 16) ? 8 : 0;
#pragma unroll
            for (int r = 0; r < 8; ++r) {
                pw[(mb + r) * PLD + n]      = (__bf16)s0[r];
                pw[(mb + r) * PLD + 16 + n] = (__bf16)s1[r];
            }
            asm volatile("s_wait_dscnt 0x0" ::: "memory");
            const v16bf pa = lds_a_frag(pw, PLD);
#pragma unroll
            for (int c = 0; c < 4; ++c)
                o[c] = wmma_bf16(pa, lds_bk_frag(v_sm + c * 16, KLD), o[c]);
        }
    }

    // ---- normalize and write out [B*N, 512] with head cols at h*64 ----
    const int n  = lane & 15;
    const int mb = (lane & 16) ? 8 : 0;
#pragma unroll
    for (int r = 0; r < 8; ++r) {
        const float invl = 1.0f / l_i[r];
        const size_t row = (size_t)(b * NPATCH + q0 + wave * 16 + mb + r);
#pragma unroll
        for (int c = 0; c < 4; ++c)
            outp[row * DIM + h * HD + c * 16 + n] = o[c][r] * invl;
    }
}

// ===========================================================================
// per-batch means -> concat [8, 1024]
// ===========================================================================
__global__ __launch_bounds__(256)
void k_means(const float* __restrict__ spec, const float* __restrict__ spat,
             float* __restrict__ concat)
{
    const int b = blockIdx.x;
    for (int c = threadIdx.x; c < DIM; c += 256) {
        float s1 = 0.f, s2 = 0.f;
        for (int nn = 0; nn < NPATCH; ++nn) {
            const size_t idx = (size_t)(b * NPATCH + nn) * DIM + c;
            s1 += spec[idx]; s2 += spat[idx];
        }
        concat[b * 1024 + c]       = s1 * (1.0f / NPATCH);
        concat[b * 1024 + DIM + c] = s2 * (1.0f / NPATCH);
    }
}

// ===========================================================================
// gating head: gate[b,0..2] = softmax(gelu(LN(concat@w1+b1))@w2+b2)
// ===========================================================================
__global__ __launch_bounds__(256)
void k_gate(const float* __restrict__ concat,
            const float* __restrict__ w1, const float* __restrict__ b1,
            const float* __restrict__ lng, const float* __restrict__ lnb,
            const float* __restrict__ w2, const float* __restrict__ b2,
            float* __restrict__ gate)
{
    __shared__ float cb[1024];
    __shared__ float gb[512];
    __shared__ float red[16];
    const int b = blockIdx.x, tid = threadIdx.x;
    const int wave = tid >> 5, lane = tid & 31;

    for (int i = tid; i < 1024; i += 256) cb[i] = concat[b * 1024 + i];
    __syncthreads();

    float y[2];
#pragma unroll
    for (int t = 0; t < 2; ++t) {
        const int cc = tid + t * 256;
        float acc = b1[cc];
        for (int k = 0; k < 1024; ++k) acc += cb[k] * w1[k * 512 + cc];
        y[t] = acc;
    }
    float s = y[0] + y[1], s2 = y[0] * y[0] + y[1] * y[1];
#pragma unroll
    for (int off = 1; off < 32; off <<= 1) {
        s  += __shfl_xor(s,  off, 32);
        s2 += __shfl_xor(s2, off, 32);
    }
    if (lane == 0) { red[wave] = s; red[8 + wave] = s2; }
    __syncthreads();
    s = 0.f; s2 = 0.f;
#pragma unroll
    for (int w = 0; w < 8; ++w) { s += red[w]; s2 += red[8 + w]; }
    const float mu  = s * (1.0f / 512.0f);
    const float var = fmaxf(s2 * (1.0f / 512.0f) - mu * mu, 0.0f);
    const float rs  = rsqrtf(var + 1e-5f);
#pragma unroll
    for (int t = 0; t < 2; ++t) {
        const int cc = tid + t * 256;
        gb[cc] = gelu_f((y[t] - mu) * rs * lng[cc] + lnb[cc]);
    }
    __syncthreads();
    if (tid < 3) {
        float z = b2[tid];
        for (int k = 0; k < 512; ++k) z += gb[k] * w2[k * 3 + tid];
        red[tid] = z;
    }
    __syncthreads();
    if (tid == 0) {
        const float z0 = red[0], z1 = red[1], z2 = red[2];
        const float mx = fmaxf(z0, fmaxf(z1, z2));
        const float e0 = expf(z0 - mx), e1 = expf(z1 - mx), e2 = expf(z2 - mx);
        const float inv = 1.0f / (e0 + e1 + e2);
        gate[b * 3 + 0] = e0 * inv;
        gate[b * 3 + 1] = e1 * inv;
        gate[b * 3 + 2] = e2 * inv;
    }
}

// ===========================================================================
// elementwise kernels
// ===========================================================================
__global__ __launch_bounds__(256)
void k_add2(const float* __restrict__ a, const float* __restrict__ b,
            float* __restrict__ o, int n)
{
    const int i = blockIdx.x * 256 + threadIdx.x;
    if (i < n) o[i] = a[i] + b[i];
}

__global__ __launch_bounds__(256)
void k_fuse(const float* __restrict__ spec, const float* __restrict__ spat,
            const float* __restrict__ vol, const float* __restrict__ gate,
            float* __restrict__ o, int n)
{
    const int i = blockIdx.x * 256 + threadIdx.x;
    if (i < n) {
        const int b = i >> 19;   // NPATCH*DIM = 2^19 per batch
        o[i] = gate[b * 3 + 0] * spec[i] + gate[b * 3 + 1] * spat[i]
             + gate[b * 3 + 2] * vol[i];
    }
}

__global__ __launch_bounds__(256)
void k_final(const float* __restrict__ fused, const float* __restrict__ fr,
             const float* __restrict__ hsi, const float* __restrict__ lidar,
             const float* __restrict__ alpha, const float* __restrict__ beta,
             float* __restrict__ out, int n)
{
    const int i = blockIdx.x * 256 + threadIdx.x;
    if (i < n)
        out[i] = fused[i] + alpha[0] * fr[i]
               + 0.5f * beta[0] * (hsi[i] + lidar[i]);
}

// ===========================================================================
// host-side orchestration
// ===========================================================================
extern "C" void kernel_launch(void* const* d_in, const int* in_sizes, int n_in,
                              void* d_out, int out_size, void* d_ws, size_t ws_size,
                              hipStream_t stream)
{
    const float* hsi   = (const float*)d_in[0];
    const float* lidar = (const float*)d_in[1];
    const float* temp  = (const float*)d_in[2];
    const float* sq_w  = (const float*)d_in[3];
    const float* sq_b  = (const float*)d_in[4];
    const float* sq_g  = (const float*)d_in[5];
    const float* sq_lb = (const float*)d_in[6];
    const float* spw   = (const float*)d_in[7];
    const float* spb   = (const float*)d_in[8];
    const float* pq_w  = (const float*)d_in[9];
    const float* pq_b  = (const float*)d_in[10];
    const float* pq_g  = (const float*)d_in[11];
    const float* pq_lb = (const float*)d_in[12];
    const float* ppw   = (const float*)d_in[13];
    const float* ppb   = (const float*)d_in[14];
    const float* vfw1  = (const float*)d_in[15];
    const float* vfb1  = (const float*)d_in[16];
    const float* vfw2  = (const float*)d_in[17];
    const float* vfb2  = (const float*)d_in[18];
    const float* agw1  = (const float*)d_in[19];
    const float* agb1  = (const float*)d_in[20];
    const float* aglg  = (const float*)d_in[21];
    const float* aglb  = (const float*)d_in[22];
    const float* agw2  = (const float*)d_in[23];
    const float* agb2  = (const float*)d_in[24];
    const float* frw1  = (const float*)d_in[25];
    const float* frb1  = (const float*)d_in[26];
    const float* frw2  = (const float*)d_in[27];
    const float* frb2  = (const float*)d_in[28];
    const float* alpha = (const float*)d_in[29];
    const float* beta  = (const float*)d_in[30];
    float* out = (float*)d_out;
    float* ws  = (float*)d_ws;

    // workspace layout (floats) — branches processed sequentially for reuse
    const size_t A_OFF  = 0;                       // 8192*1536  (qkv / h1 / h2+fr_out)
    const size_t B_OFF  = (size_t)MROWS * TD;      // +4.19M     (attn out / volumetric)
    const size_t S_OFF  = B_OFF + (size_t)MROWS * DIM;   // out_spectral
    const size_t P_OFF  = S_OFF + (size_t)MROWS * DIM;   // out_spatial
    const size_t C_OFF  = P_OFF + (size_t)MROWS * DIM;   // s_sum -> fused
    const size_t CAT_OFF  = C_OFF + (size_t)MROWS * DIM; // concat [8,1024]
    const size_t GATE_OFF = CAT_OFF + 8 * 1024;          // gate [8,3]
    const size_t FR_OFF = A_OFF + (size_t)MROWS * 1024;  // fr_out in tail of A

    const int NEL = MROWS * DIM;           // 4,194,304
    const dim3 blk(256);
    const dim3 gEw((NEL + 255) / 256);
    auto gGemm = [](int Nc) { return dim3(Nc / 128, MROWS / 128); };

    // ---- spectral branch ----
    k_gemm<0><<<gGemm(TD), blk, 0, stream>>>(hsi, sq_w, sq_b, ws + A_OFF, DIM, TD);
    k_ln_rows<<<MROWS, blk, 0, stream>>>(ws + A_OFF, sq_g, sq_lb, TD);
    k_attn<<<512, blk, 0, stream>>>(ws + A_OFF, temp, ws + B_OFF);
    k_gemm<0><<<gGemm(DIM), blk, 0, stream>>>(ws + B_OFF, spw, spb, ws + S_OFF, DIM, DIM);

    // ---- spatial branch ----
    k_gemm<0><<<gGemm(TD), blk, 0, stream>>>(lidar, pq_w, pq_b, ws + A_OFF, DIM, TD);
    k_ln_rows<<<MROWS, blk, 0, stream>>>(ws + A_OFF, pq_g, pq_lb, TD);
    k_attn<<<512, blk, 0, stream>>>(ws + A_OFF, temp, ws + B_OFF);
    k_gemm<0><<<gGemm(DIM), blk, 0, stream>>>(ws + B_OFF, ppw, ppb, ws + P_OFF, DIM, DIM);

    // ---- volumetric fusion MLP ----
    k_add2<<<gEw, blk, 0, stream>>>(ws + S_OFF, ws + P_OFF, ws + C_OFF, NEL);
    k_gemm<1><<<gGemm(1024), blk, 0, stream>>>(ws + C_OFF, vfw1, vfb1, ws + A_OFF, DIM, 1024);
    k_gemm<0><<<gGemm(DIM), blk, 0, stream>>>(ws + A_OFF, vfw2, vfb2, ws + B_OFF, 1024, DIM);

    // ---- adaptive gating ----
    k_means<<<8, blk, 0, stream>>>(ws + S_OFF, ws + P_OFF, ws + CAT_OFF);
    k_gate<<<8, blk, 0, stream>>>(ws + CAT_OFF, agw1, agb1, aglg, aglb, agw2, agb2,
                                  ws + GATE_OFF);
    k_fuse<<<gEw, blk, 0, stream>>>(ws + S_OFF, ws + P_OFF, ws + B_OFF,
                                    ws + GATE_OFF, ws + C_OFF, NEL);

    // ---- recalibration MLP + final combine ----
    k_gemm<1><<<gGemm(1024), blk, 0, stream>>>(ws + C_OFF, frw1, frb1, ws + A_OFF, DIM, 1024);
    k_gemm<0><<<gGemm(DIM), blk, 0, stream>>>(ws + A_OFF, frw2, frb2, ws + FR_OFF, 1024, DIM);
    k_final<<<gEw, blk, 0, stream>>>(ws + C_OFF, ws + FR_OFF, hsi, lidar,
                                     alpha, beta, out, NEL);
}
